// SkeletonConv_9474697855040
// MI455X (gfx1250) — matrix operand order repaired and checked by reading, not compiled, not run on hardware
//
#include <hip/hip_runtime.h>
#include <hip/hip_bf16.h>

// EdgeConv x3 (DGCNN block) for MI455X / gfx1250, wave32 + WMMA.
//
// Restructure: W @ [neigh - center ; center] == Wa @ x[idx] + (Wb - Wa) @ x.
// Per layer: one f16 GEMM  H[p, 0:2O] = Xt[p, 0:C] * Wp[C, 2O]  (f32 accum),
// then gather/bias/leakyrelu/max epilogue. Cuts FLOPs 4x (K=4) vs reference.
//
// GEMM waves own a 32x64 output tile (2 M-tiles x 4 N-tiles, 8 WMMA accums):
// B fragments are reused across both M-tiles -> half
// the weight-panel L2 traffic vs a 16x64 tile.
//
// Workspace layout (needs ~152 MB):
//   [0,        48 MiB)  Xt   : f16 [B*N][Cin]   (rewritten per layer)
//   [48 MiB,  144 MiB)  H    : f32 [B*N][2O]
//   [144 MiB, 145 MiB)  Wp   : f16 [Cin][2O]    (Wp[c][o]=W[o][c]; Wp[c][O+o]=W[o][C+c]-W[o][c])

typedef __attribute__((ext_vector_type(16))) _Float16 v16h;
typedef __attribute__((ext_vector_type(8)))  _Float16 v8h;
typedef __attribute__((ext_vector_type(8)))  float    v8f;

constexpr int BATCH = 2048;
constexpr int NPT   = 24;     // points per cloud
constexpr int KNN   = 4;
constexpr int NPts  = BATCH * NPT;        // 49152 points = 1536 M-tile-pairs of 32
constexpr int OTOT  = 448;                // 256 + 128 + 64

constexpr size_t XT_OFF = 0;              // f16, max 49152*512*2 = 48 MiB
constexpr size_t H_OFF  = 50331648;       // f32, max 49152*512*4 = 96 MiB
constexpr size_t WP_OFF = 150994944;      // f16, max 512*512*2   = 0.5 MiB

// ---------------------------------------------------------------------------
// Pack x [B][C][N] f32  ->  Xt [B*N][C] f16.  One block per batch element:
// its 48 KB slice becomes cache-resident, writes are coalesced.
__global__ __launch_bounds__(256) void pack_x_kernel(const float* __restrict__ x,
                                                     _Float16* __restrict__ xt) {
  const int b = blockIdx.x;
  const float* xb = x + (size_t)b * 512 * NPT;
  _Float16* xtb = xt + (size_t)b * NPT * 512;
  for (int e = threadIdx.x; e < 512 * NPT; e += 256) {
    int c = e & 511;           // fastest -> coalesced f16 writes
    int n = e >> 9;            // 0..23
    xtb[(size_t)n * 512 + c] = (_Float16)xb[(size_t)c * NPT + n];
  }
}

// Pack W [O][2C] f32 -> Wp [Cin][2O] f16 with the Wa / (Wb-Wa) split.
__global__ __launch_bounds__(256) void pack_w_kernel(const float* __restrict__ W,
                                                     _Float16* __restrict__ wp,
                                                     int Cin, int O) {
  const int twoO = 2 * O;
  int t = blockIdx.x * 256 + threadIdx.x;
  if (t >= Cin * twoO) return;
  int j = t % twoO;            // fastest -> coalesced writes
  int c = t / twoO;
  float v;
  if (j < O) {
    v = W[(size_t)j * (2 * Cin) + c];                                  // Wa
  } else {
    int o = j - O;
    v = W[(size_t)o * (2 * Cin) + Cin + c] - W[(size_t)o * (2 * Cin) + c]; // Wb-Wa
  }
  wp[(size_t)c * twoO + j] = (_Float16)v;
}

// ---------------------------------------------------------------------------
// WMMA GEMM: H[NPts][twoO] = Xt[NPts][Cin] * Wp[Cin][twoO], f16 in / f32 out.
// Block = 4 waves; wave w owns 32 points (2 M-tiles) x 64 out-channels
// (4 N-tiles): 8 accumulators, A reused 4x, B reused 2x.
__global__ __launch_bounds__(128) void gemm_wmma_kernel(const _Float16* __restrict__ XT,
                                                        const _Float16* __restrict__ WP,
                                                        float* __restrict__ H,
                                                        int Cin, int twoO) {
  const int lane = threadIdx.x & 31;
  const int wave = threadIdx.x >> 5;
  const int p0 = (blockIdx.x * 4 + wave) * 32;   // 32-point tile base
  const int n0 = blockIdx.y * 64;                // out-channel strip base

  const int m    = lane & 15;
  const int half = lane >> 4;                    // 0: K 0-7/16-23, 1: K 8-15/24-31

  // A: row-major f16 [point][Cin]; ISA 16x32 A layout per lane.
  const _Float16* aptr0 = XT + (size_t)(p0 + m) * Cin + half * 8;
  const _Float16* aptr1 = XT + (size_t)(p0 + 16 + m) * Cin + half * 8;
  // B: row-major f16 [Cin][twoO]; lane holds column N = lane&15, K in table order.
  const _Float16* bbase = WP + (size_t)(half * 8) * twoO + n0 + m;

  v8f acc[8];
#pragma unroll
  for (int i = 0; i < 8; ++i) acc[i] = v8f{0.f, 0.f, 0.f, 0.f, 0.f, 0.f, 0.f, 0.f};

  for (int k0 = 0; k0 < Cin; k0 += 32) {
    // A fragments: two 16-byte loads per lane per M-tile.
    v8h a0lo = *(const v8h*)(aptr0 + k0);        // K = k0 + 8*half + [0..7]
    v8h a0hi = *(const v8h*)(aptr0 + k0 + 16);   // K = k0 + 16 + 8*half + [0..7]
    v8h a1lo = *(const v8h*)(aptr1 + k0);
    v8h a1hi = *(const v8h*)(aptr1 + k0 + 16);
    v16h a0 = __builtin_shufflevector(a0lo, a0hi, 0, 1, 2, 3, 4, 5, 6, 7,
                                      8, 9, 10, 11, 12, 13, 14, 15);
    v16h a1 = __builtin_shufflevector(a1lo, a1hi, 0, 1, 2, 3, 4, 5, 6, 7,
                                      8, 9, 10, 11, 12, 13, 14, 15);
    __builtin_prefetch(aptr0 + k0 + 32, 0, 1);   // global_prefetch_b8 next K step
    __builtin_prefetch(aptr1 + k0 + 32, 0, 1);

    const _Float16* bk = bbase + (size_t)k0 * twoO;  // element s: K = k0 + 8*half + s + (s&8)
    v16h b0, b1, b2, b3;
#pragma unroll
    for (int s = 0; s < 16; ++s) {
      size_t r = (size_t)(s + (s & 8)) * twoO;
      b0[s] = bk[r];
      b1[s] = bk[r + 16];
      b2[s] = bk[r + 32];
      b3[s] = bk[r + 48];
    }

    acc[0] = __builtin_amdgcn_wmma_f32_16x16x32_f16(false, a0, false, b0,
                                                    (short)0, acc[0], false, false);
    acc[1] = __builtin_amdgcn_wmma_f32_16x16x32_f16(false, a0, false, b1,
                                                    (short)0, acc[1], false, false);
    acc[2] = __builtin_amdgcn_wmma_f32_16x16x32_f16(false, a0, false, b2,
                                                    (short)0, acc[2], false, false);
    acc[3] = __builtin_amdgcn_wmma_f32_16x16x32_f16(false, a0, false, b3,
                                                    (short)0, acc[3], false, false);
    acc[4] = __builtin_amdgcn_wmma_f32_16x16x32_f16(false, a1, false, b0,
                                                    (short)0, acc[4], false, false);
    acc[5] = __builtin_amdgcn_wmma_f32_16x16x32_f16(false, a1, false, b1,
                                                    (short)0, acc[5], false, false);
    acc[6] = __builtin_amdgcn_wmma_f32_16x16x32_f16(false, a1, false, b2,
                                                    (short)0, acc[6], false, false);
    acc[7] = __builtin_amdgcn_wmma_f32_16x16x32_f16(false, a1, false, b3,
                                                    (short)0, acc[7], false, false);
  }

  // C/D layout: VGPR r -> M = 8*half + r, lane&15 -> N.
  const int nl = lane & 15;
#pragma unroll
  for (int t = 0; t < 2; ++t) {
#pragma unroll
    for (int j = 0; j < 4; ++j) {
#pragma unroll
      for (int r = 0; r < 8; ++r) {
        H[(size_t)(p0 + t * 16 + half * 8 + r) * twoO + n0 + j * 16 + nl] =
            acc[t * 4 + j][r];
      }
    }
  }
}

// ---------------------------------------------------------------------------
// Epilogue: h[b,n,k,o] = Ya[b, idx[n,k], o] + Yd[b, n, o] + bias[o];
// out = max_k leakyrelu(h).  Writes fp32 output slice and next layer's f16 Xt.
__global__ __launch_bounds__(256) void combine_kernel(const float* __restrict__ H,
                                                      const int* __restrict__ idx,
                                                      const float* __restrict__ bias,
                                                      float* __restrict__ out,
                                                      _Float16* __restrict__ xt_next,
                                                      int O, int off) {
  int t = blockIdx.x * 256 + threadIdx.x;
  const int total = BATCH * NPT * O;
  if (t >= total) return;
  const int o = t % O;                  // fastest -> coalesced H reads
  const int n = (t / O) % NPT;
  const int b = t / (O * NPT);
  const int twoO = 2 * O;

  const float* Hb = H + (size_t)b * NPT * twoO;
  const float yd = Hb[(size_t)n * twoO + O + o] + bias[o];

  float mx = -3.4e38f;
#pragma unroll
  for (int k = 0; k < KNN; ++k) {
    int nn = idx[n * KNN + k];
    float h = Hb[(size_t)nn * twoO + o] + yd;
    h = (h > 0.f) ? h : 0.2f * h;       // leaky relu, slope 0.2
    mx = fmaxf(mx, h);
  }

  out[(size_t)b * OTOT * NPT + (size_t)(off + o) * NPT + n] = mx;
  if (xt_next) xt_next[((size_t)b * NPT + n) * O + o] = (_Float16)mx;
}

// ---------------------------------------------------------------------------
extern "C" void kernel_launch(void* const* d_in, const int* in_sizes, int n_in,
                              void* d_out, int out_size, void* d_ws, size_t ws_size,
                              hipStream_t stream) {
  const float* x  = (const float*)d_in[0];
  const int* idx  = (const int*)d_in[1];
  const float* W1 = (const float*)d_in[2];
  const float* b1 = (const float*)d_in[3];
  const float* W2 = (const float*)d_in[4];
  const float* b2 = (const float*)d_in[5];
  const float* W3 = (const float*)d_in[6];
  const float* b3 = (const float*)d_in[7];
  float* out = (float*)d_out;

  char* ws = (char*)d_ws;
  _Float16* XT = (_Float16*)(ws + XT_OFF);
  float*    H  = (float*)(ws + H_OFF);
  _Float16* WP = (_Float16*)(ws + WP_OFF);

  // Pack x -> f16 point-major.
  pack_x_kernel<<<BATCH, 256, 0, stream>>>(x, XT);

  const int mblocks = NPts / 128;  // 384 blocks x 4 waves x 32 points = 49152

  // ---- Layer 1: Cin=512, O=256 ----
  pack_w_kernel<<<(512 * 512 + 255) / 256, 256, 0, stream>>>(W1, WP, 512, 256);
  gemm_wmma_kernel<<<dim3(mblocks, 512 / 64), 128, 0, stream>>>(XT, WP, H, 512, 512);
  combine_kernel<<<(BATCH * NPT * 256 + 255) / 256, 256, 0, stream>>>(
      H, idx, b1, out, XT, 256, 0);

  // ---- Layer 2: Cin=256, O=128 ----
  pack_w_kernel<<<(256 * 256 + 255) / 256, 256, 0, stream>>>(W2, WP, 256, 128);
  gemm_wmma_kernel<<<dim3(mblocks, 256 / 64), 128, 0, stream>>>(XT, WP, H, 256, 256);
  combine_kernel<<<(BATCH * NPT * 128 + 255) / 256, 256, 0, stream>>>(
      H, idx, b2, out, XT, 128, 256);

  // ---- Layer 3: Cin=128, O=64 ----
  pack_w_kernel<<<(128 * 128 + 255) / 256, 256, 0, stream>>>(W3, WP, 128, 64);
  gemm_wmma_kernel<<<dim3(mblocks, 128 / 64), 128, 0, stream>>>(XT, WP, H, 128, 128);
  combine_kernel<<<(BATCH * NPT * 64 + 255) / 256, 256, 0, stream>>>(
      H, idx, b3, out, nullptr, 64, 384);
}
